// MultiHeadAttentionBlock_3779571221048
// MI455X (gfx1250) — matrix operand order repaired and checked by reading, not compiled, not run on hardware
//
#include <hip/hip_runtime.h>
#include <hip/hip_bf16.h>

typedef __attribute__((ext_vector_type(16))) _Float16 v16h;
typedef __attribute__((ext_vector_type(8)))  float    v8f;

#define WMMA_F16(a, b, c) \
  __builtin_amdgcn_wmma_f32_16x16x32_f16(false, (a), false, (b), (short)0, (c), false, false)

// Problem constants
static constexpr int Bc = 2, Sc = 2048, Dc = 768, Hc = 12, DKc = 64;
static constexpr int Mtot = Bc * Sc;          // 4096 rows
static constexpr float NEGC = -10000.0f;

// ---------------------------------------------------------------------------
// f32 -> f16 elementwise conversion
// ---------------------------------------------------------------------------
__global__ void cvt_f32_f16(const float* __restrict__ s, _Float16* __restrict__ d, int n) {
  int i = blockIdx.x * blockDim.x + threadIdx.x;
  if (i < n) d[i] = (_Float16)s[i];
}

// ---------------------------------------------------------------------------
// WMMA fragment loaders (CDNA5 wave32 layouts, CDNA5 ISA §7.12.2)
// A 16x32 (MxK): lane<16 -> row M=lane, halves {K0..7, K16..23};
//                lane>=16 -> row M=lane-16, halves {K8..15, K24..31}
// ---------------------------------------------------------------------------
__device__ inline v16h load_a16(const _Float16* __restrict__ p0, int ld, int lane) {
  int l = lane & 15, hi = lane >> 4;
  const _Float16* p = p0 + (size_t)l * ld + hi * 8;
  v16h a;
#pragma unroll
  for (int i = 0; i < 8; ++i) { a[i] = p[i]; a[8 + i] = p[16 + i]; }
  return a;
}

// B 32x16 (KxN): lane<16 -> col N=lane, {K0..7, K16..23}; lane>=16 -> col N=lane-16, {K8..15, K24..31}
// p0 must point at element [k0][n0] of a matrix stored as [N][K] (K contiguous).
__device__ inline v16h load_b16(const _Float16* __restrict__ p0, int ld, int lane) {
  int l = lane & 15, hi = lane >> 4;
  const _Float16* p = p0 + (size_t)l * ld + hi * 8;
  v16h b;
#pragma unroll
  for (int i = 0; i < 8; ++i) { b[i] = p[i]; b[8 + i] = p[16 + i]; }
  return b;
}

// ---------------------------------------------------------------------------
// Fused QKV projection: C[m,n] = sum_k X[m,k] * W[n,k] + bias[n]
// Wave computes a 16x64 tile. Q,K stored [B,H,S,DK]; V stored transposed [B,H,DK,S].
// tasks: which(3) x mt(256) x nt(12) = 9216 waves -> 1152 blocks of 8 waves
// ---------------------------------------------------------------------------
__global__ __launch_bounds__(256) void qkv_gemm(
    const _Float16* __restrict__ x16,   // [3][4096][768]
    const _Float16* __restrict__ w16,   // [3][768][768]
    const float* __restrict__ bq, const float* __restrict__ bk, const float* __restrict__ bv,
    _Float16* __restrict__ qh, _Float16* __restrict__ kh, _Float16* __restrict__ vt) {
  int lane = threadIdx.x & 31, wave = threadIdx.x >> 5;
  int t = blockIdx.x * 8 + wave;
  int which = t / 3072; t %= 3072;
  int mt = t / 12, nt = t % 12;
  int m0 = mt * 16, n0 = nt * 64;
  const _Float16* X = x16 + (size_t)which * Mtot * Dc;
  const _Float16* W = w16 + (size_t)which * Dc * Dc;
  const float* bias = (which == 0) ? bq : (which == 1) ? bk : bv;

  v8f c[4] = {};
  for (int k0 = 0; k0 < Dc; k0 += 32) {
    v16h a = load_a16(X + (size_t)m0 * Dc + k0, Dc, lane);
#pragma unroll
    for (int s = 0; s < 4; ++s) {
      v16h b = load_b16(W + (size_t)(n0 + s * 16) * Dc + k0, Dc, lane);
      c[s] = WMMA_F16(a, b, c[s]);
    }
  }
  int l = lane & 15, hi = lane >> 4;
  int h = n0 / DKc;        // 64-wide n-tile == one head
  int b_ = m0 / Sc, s0 = m0 % Sc;
#pragma unroll
  for (int s = 0; s < 4; ++s) {
    int d = s * 16 + l;                     // d within head
    float bb = bias[n0 + d];
#pragma unroll
    for (int r = 0; r < 8; ++r) {
      int row = s0 + r + hi * 8;
      _Float16 val = (_Float16)(c[s][r] + bb);
      if (which == 0)
        qh[(((size_t)b_ * Hc + h) * Sc + row) * DKc + d] = val;
      else if (which == 1)
        kh[(((size_t)b_ * Hc + h) * Sc + row) * DKc + d] = val;
      else
        vt[(((size_t)b_ * Hc + h) * DKc + d) * Sc + row] = val;   // transposed
    }
  }
}

// ---------------------------------------------------------------------------
// Causal flash attention. grid = (S/16/8, B*H), block = 256 (8 waves).
// Each wave owns 16 query rows; loops kv in steps of 32 up to the diagonal.
// ---------------------------------------------------------------------------
__global__ __launch_bounds__(256) void attn_kernel(
    const _Float16* __restrict__ qh, const _Float16* __restrict__ kh,
    const _Float16* __restrict__ vt, _Float16* __restrict__ attnout) {
  __shared__ _Float16 ldsP[8][16][32];     // per-wave 1KB P transpose staging
  int lane = threadIdx.x & 31, wave = threadIdx.x >> 5;
  int l = lane & 15, hi = lane >> 4;
  int bh = blockIdx.y;
  int b_ = bh / Hc, h = bh % Hc;
  int m0 = (blockIdx.x * 8 + wave) * 16;

  const _Float16* Qb = qh + (size_t)bh * Sc * DKc;
  const _Float16* Kb = kh + (size_t)bh * Sc * DKc;
  const _Float16* Vb = vt + (size_t)bh * DKc * Sc;

  v16h qf0 = load_a16(Qb + (size_t)m0 * DKc + 0,  DKc, lane);
  v16h qf1 = load_a16(Qb + (size_t)m0 * DKc + 32, DKc, lane);

  v8f o[4] = {};
  float mrow[8], lrow[8];
#pragma unroll
  for (int r = 0; r < 8; ++r) { mrow[r] = -1e30f; lrow[r] = 0.f; }

  int nsteps = (m0 + 16 + 31) >> 5;        // only tiles touching the causal band
  for (int j = 0; j < nsteps; ++j) {
    int kv0 = j * 32;
    // prefetch next step's K (32 rows) and V (64 rows) cachelines one iter ahead
    // (emits global_prefetch_b8; covers the serial softmax->wmma bubble)
    if (j + 1 < nsteps) {
      int kvn = kv0 + 32;
      __builtin_prefetch(Kb + (size_t)(kvn + lane) * DKc, 0, 3);
      __builtin_prefetch(Vb + (size_t)lane * Sc + kvn, 0, 3);
      __builtin_prefetch(Vb + (size_t)(lane + 32) * Sc + kvn, 0, 3);
    }
    // scores: two 16x16 tiles over 32 kv columns, K-dim = DK = 64 (two wmma each)
    v8f sc[2];
#pragma unroll
    for (int sub = 0; sub < 2; ++sub) {
      v16h bk0 = load_b16(Kb + (size_t)(kv0 + sub * 16) * DKc + 0,  DKc, lane);
      v16h bk1 = load_b16(Kb + (size_t)(kv0 + sub * 16) * DKc + 32, DKc, lane);
      v8f acc = {};
      acc = WMMA_F16(qf0, bk0, acc);
      acc = WMMA_F16(qf1, bk1, acc);
      sc[sub] = acc;
    }
    // scale + causal mask (mask==0 <=> kv > q -> NEG), like reference
    float p[2][8];
#pragma unroll
    for (int sub = 0; sub < 2; ++sub) {
      int n = kv0 + sub * 16 + l;
#pragma unroll
      for (int r = 0; r < 8; ++r) {
        int m = m0 + r + hi * 8;
        float v = sc[sub][r] * 0.125f;     // 1/sqrt(64)
        p[sub][r] = (n > m) ? NEGC : v;
      }
    }
    // row max across 16 lanes of each half (rows 0-7 in lanes 0-15, 8-15 in 16-31)
    float rm[8];
#pragma unroll
    for (int r = 0; r < 8; ++r) {
      float v = fmaxf(p[0][r], p[1][r]);
      v = fmaxf(v, __shfl_xor(v, 1, 32));
      v = fmaxf(v, __shfl_xor(v, 2, 32));
      v = fmaxf(v, __shfl_xor(v, 4, 32));
      v = fmaxf(v, __shfl_xor(v, 8, 32));
      rm[r] = v;
    }
    float alpha[8];
#pragma unroll
    for (int r = 0; r < 8; ++r) {
      float mn = fmaxf(mrow[r], rm[r]);
      alpha[r] = __expf(mrow[r] - mn);
      mrow[r] = mn;
    }
#pragma unroll
    for (int sub = 0; sub < 2; ++sub)
#pragma unroll
      for (int r = 0; r < 8; ++r) p[sub][r] = __expf(p[sub][r] - mrow[r]);
    // row sum + state update
#pragma unroll
    for (int r = 0; r < 8; ++r) {
      float v = p[0][r] + p[1][r];
      v += __shfl_xor(v, 1, 32);
      v += __shfl_xor(v, 2, 32);
      v += __shfl_xor(v, 4, 32);
      v += __shfl_xor(v, 8, 32);
      lrow[r] = lrow[r] * alpha[r] + v;
    }
#pragma unroll
    for (int d = 0; d < 4; ++d)
#pragma unroll
      for (int r = 0; r < 8; ++r) o[d][r] *= alpha[r];
    // P: C-layout -> A-layout via per-wave LDS slice (same-wave DS ops in-order)
#pragma unroll
    for (int sub = 0; sub < 2; ++sub)
#pragma unroll
      for (int r = 0; r < 8; ++r)
        ldsP[wave][r + hi * 8][sub * 16 + l] = (_Float16)p[sub][r];
    v16h pf;
    {
      int c0 = hi * 8;
#pragma unroll
      for (int i = 0; i < 8; ++i) {
        pf[i]     = ldsP[wave][l][c0 + i];
        pf[8 + i] = ldsP[wave][l][c0 + 16 + i];
      }
    }
    // O += P @ V : B fragment from transposed V [DK][S] (contiguous kv per lane)
#pragma unroll
    for (int d = 0; d < 4; ++d) {
      v16h bvv = load_b16(Vb + (size_t)(d * 16) * Sc + kv0, Sc, lane);
      o[d] = WMMA_F16(pf, bvv, o[d]);
    }
  }
  // normalize and emit [B*S, D] f16 rows (heads re-interleaved)
#pragma unroll
  for (int r = 0; r < 8; ++r) lrow[r] = 1.f / lrow[r];
#pragma unroll
  for (int d = 0; d < 4; ++d)
#pragma unroll
    for (int r = 0; r < 8; ++r) {
      int row = m0 + r + hi * 8;
      attnout[((size_t)(b_ * Sc + row)) * Dc + h * DKc + d * 16 + l] =
          (_Float16)(o[d][r] * lrow[r]);
    }
}

// ---------------------------------------------------------------------------
// Output projection: out[m,n] = sum_k A[m,k] * Wo[n,k] + bo[n]  (fp32 out)
// tasks: 256 x 12 = 3072 waves -> 384 blocks of 8
// ---------------------------------------------------------------------------
__global__ __launch_bounds__(256) void oproj_gemm(
    const _Float16* __restrict__ A, const _Float16* __restrict__ Wo16,
    const float* __restrict__ bo, float* __restrict__ out) {
  int lane = threadIdx.x & 31, wave = threadIdx.x >> 5;
  int t = blockIdx.x * 8 + wave;
  int mt = t / 12, nt = t % 12;
  int m0 = mt * 16, n0 = nt * 64;
  v8f c[4] = {};
  for (int k0 = 0; k0 < Dc; k0 += 32) {
    v16h a = load_a16(A + (size_t)m0 * Dc + k0, Dc, lane);
#pragma unroll
    for (int s = 0; s < 4; ++s) {
      v16h b = load_b16(Wo16 + (size_t)(n0 + s * 16) * Dc + k0, Dc, lane);
      c[s] = WMMA_F16(a, b, c[s]);
    }
  }
  int l = lane & 15, hi = lane >> 4;
#pragma unroll
  for (int s = 0; s < 4; ++s) {
    float bb = bo[n0 + s * 16 + l];
#pragma unroll
    for (int r = 0; r < 8; ++r)
      out[(size_t)(m0 + r + hi * 8) * Dc + n0 + s * 16 + l] = c[s][r] + bb;
  }
}

// ---------------------------------------------------------------------------
extern "C" void kernel_launch(void* const* d_in, const int* in_sizes, int n_in,
                              void* d_out, int out_size, void* d_ws, size_t ws_size,
                              hipStream_t stream) {
  const float* query = (const float*)d_in[0];
  const float* key_  = (const float*)d_in[1];
  const float* value = (const float*)d_in[2];
  // d_in[3] = causal mask (int32) — causality is computed analytically
  const float* Wq = (const float*)d_in[4];  const float* bq = (const float*)d_in[5];
  const float* Wk = (const float*)d_in[6];  const float* bk = (const float*)d_in[7];
  const float* Wv = (const float*)d_in[8];  const float* bv = (const float*)d_in[9];
  const float* Wo = (const float*)d_in[10]; const float* bo = (const float*)d_in[11];
  float* out = (float*)d_out;
  (void)in_sizes; (void)n_in; (void)out_size; (void)ws_size;

  const size_t nX  = (size_t)Mtot * Dc;     // 4096*768
  const size_t nW  = (size_t)Dc * Dc;       // 768*768
  const size_t nHd = (size_t)Bc * Hc * Sc * DKc;

  // Workspace layout (bytes). x16 is dead after qkv_gemm, so the attention
  // output buffer `ao` aliases it (stream-ordered, safe). Peak ~42.5 MB.
  char* ws = (char*)d_ws;
  _Float16* x16  = (_Float16*)ws;                        // 3 * nX   (also: ao later)
  _Float16* w16  = (_Float16*)(ws + 3 * nX * 2);         // 3 * nW
  _Float16* wo16 = (_Float16*)(ws + 3 * nX * 2 + 3 * nW * 2);
  _Float16* qh   = (_Float16*)(ws + 3 * nX * 2 + 4 * nW * 2);
  _Float16* kh   = qh + nHd;
  _Float16* vt   = kh + nHd;
  _Float16* ao   = x16;                                  // alias over dead x16

  // 1) convert inputs and weights to f16
  cvt_f32_f16<<<(int)((nX + 255) / 256), 256, 0, stream>>>(query, x16,          (int)nX);
  cvt_f32_f16<<<(int)((nX + 255) / 256), 256, 0, stream>>>(key_,  x16 + nX,     (int)nX);
  cvt_f32_f16<<<(int)((nX + 255) / 256), 256, 0, stream>>>(value, x16 + 2 * nX, (int)nX);
  cvt_f32_f16<<<(int)((nW + 255) / 256), 256, 0, stream>>>(Wq, w16,          (int)nW);
  cvt_f32_f16<<<(int)((nW + 255) / 256), 256, 0, stream>>>(Wk, w16 + nW,     (int)nW);
  cvt_f32_f16<<<(int)((nW + 255) / 256), 256, 0, stream>>>(Wv, w16 + 2 * nW, (int)nW);
  cvt_f32_f16<<<(int)((nW + 255) / 256), 256, 0, stream>>>(Wo, wo16,         (int)nW);

  // 2) fused QKV projections (V stored transposed)
  qkv_gemm<<<1152, 256, 0, stream>>>(x16, w16, bq, bk, bv, qh, kh, vt);

  // 3) causal flash attention (writes ao, which aliases the now-dead x16)
  attn_kernel<<<dim3(Sc / 16 / 8, Bc * Hc), 256, 0, stream>>>(qh, kh, vt, ao);

  // 4) output projection -> fp32 d_out
  oproj_gemm<<<384, 256, 0, stream>>>(ao, wo16, bo, out);
}
